// GNNEncoder_30021821399451
// MI455X (gfx1250) — matrix operand order repaired and checked by reading, not compiled, not run on hardware
//
#include <hip/hip_runtime.h>

// ---------------------------------------------------------------------------
// 2-layer GCN encoder for MI455X (gfx1250, wave32).
//   out[d] = dinv[d] * ( sum_{s->d} g[s] + g[d] ) + b,  g = (h @ W) * dinv
// so edge propagation is a pure atomicAdd scatter (no per-edge multiply).
// GEMM1 (100000x256 @ 256x64): x pre-converted to bf16 (streaming pass, L2
// resident). Per wave: one 16-col N-tile, MBLK 16-row M-blocks; A streamed
// with clause'd global_load_b128, B from LDS-staged W1^T, 8 accumulating
// v_wmma_f32_16x16x32_bf16 per M-block. Epilogue uses float4 dinv loads,
// block-granular bounds check, and immediate-offset stores.
// ---------------------------------------------------------------------------

typedef __attribute__((ext_vector_type(16))) __bf16 v16bf;
typedef __attribute__((ext_vector_type(8)))  float  v8f;

union Frag {
  v16bf v;
  unsigned short us[16];
  unsigned int   ui[8];
};

__device__ __forceinline__ unsigned short f2bf(float f) {
  unsigned int u = __builtin_bit_cast(unsigned int, f);
  unsigned int r = u + 0x7fffu + ((u >> 16) & 1u);   // round-to-nearest-even
  return (unsigned short)(r >> 16);
}

#define FIN 256
#define HIDN 64
#define EMBN 2
#define MBLK 4   // 16-row M-blocks per wave task

// ---------------- degree / normalization ------------------------------------
__global__ void k_deg_init(float* deg, int n) {
  int i = blockIdx.x * blockDim.x + threadIdx.x;
  if (i < n) deg[i] = 1.0f;                 // self loop
}

__global__ void k_deg_count(const int* __restrict__ dst, float* deg, int e) {
  int i = blockIdx.x * blockDim.x + threadIdx.x;
  if (i < e) atomicAdd(&deg[dst[i]], 1.0f);
}

__global__ void k_dinv(float* deg, int n) {  // in-place deg -> dinv
  int i = blockIdx.x * blockDim.x + threadIdx.x;
  if (i < n) {
    float d = deg[i];
    deg[i] = (d > 0.0f) ? rsqrtf(d) : 0.0f;
  }
}

// ---------------- x (fp32) -> xbf (bf16), streaming --------------------------
__global__ void k_cvt_x(const float* __restrict__ x, unsigned int* __restrict__ xbf,
                        long nPairs) {  // nPairs = N*FIN/2
  long i = (long)blockIdx.x * blockDim.x + threadIdx.x;
  if (i < nPairs) {
    float2 v = ((const float2*)x)[i];
    xbf[i] = (unsigned int)f2bf(v.x) | ((unsigned int)f2bf(v.y) << 16);
  }
}

// ---------------- GEMM1: g1 = (xbf @ W1) * dinv ; acc1 = g1 (self loop) ------
// Block = 256 threads = 8 waves. Wave task wt = blockIdx*8+wave:
//   N-tile = wt & 3 (16 cols), M-chunk = wt >> 2 (MBLK x 16 rows).
// A layout (16-bit 16x32): lane M = l&15, K base = 8*(l>>4); v=0..3 cover
// K base+0..7 (16B run), v=4..7 cover K base+16..23 (16B run).
// B layout (16-bit 32x16): lane N = l&15, K base = 8*(l>>4); same K order as A.
__global__ void __launch_bounds__(256)
k_gemm1(const unsigned short* __restrict__ xbf, const float* __restrict__ W1,
        const float* __restrict__ dinv, float* __restrict__ g1,
        float* __restrict__ acc1, int nNodes) {
  __shared__ unsigned short ldsW[HIDN * FIN];  // W1^T bf16: [n][k], 32KB

  const int tid = threadIdx.x;
  {  // stage W1^T (bf16): ldsW[n*256+k] = bf16(W1[k*64+n])
    const int n = tid & 63;
    const int kb = (tid >> 6) * 64;
    #pragma unroll 4
    for (int j = 0; j < 64; ++j) {
      const int k = kb + j;
      ldsW[n * FIN + k] = f2bf(W1[k * HIDN + n]);
    }
  }
  __syncthreads();

  const int wave = tid >> 5;
  const int lane = tid & 31;
  const int l15  = lane & 15;
  const int lhi  = lane >> 4;

  const int wt     = blockIdx.x * 8 + wave;
  const int tile   = wt & 3;          // N-tile: cols [tile*16, tile*16+16)
  const int mbase  = (wt >> 2) * (16 * MBLK);
  if (mbase >= nNodes) return;

  // B fragments for this tile, all 8 K-steps.
  Frag bfr[8];
  {
    const unsigned short* wrow = &ldsW[(tile * 16 + l15) * FIN + lhi * 8];
    #pragma unroll
    for (int ks = 0; ks < 8; ++ks) {
      const uint4 b0 = *(const uint4*)(wrow + ks * 32);
      const uint4 b1 = *(const uint4*)(wrow + ks * 32 + 16);
      bfr[ks].ui[0] = b0.x; bfr[ks].ui[1] = b0.y;
      bfr[ks].ui[2] = b0.z; bfr[ks].ui[3] = b0.w;
      bfr[ks].ui[4] = b1.x; bfr[ks].ui[5] = b1.y;
      bfr[ks].ui[6] = b1.z; bfr[ks].ui[7] = b1.w;
    }
  }

  for (int mb = 0; mb < MBLK; ++mb) {
    const int m0 = mbase + mb * 16;
    if (m0 >= nNodes) break;

    int rowA = m0 + l15;
    if (rowA > nNodes - 1) rowA = nNodes - 1;  // clamp; stores masked later
    const unsigned short* __restrict__ xrow = xbf + (size_t)rowA * FIN + lhi * 8;

    v8f c = (v8f){0.f,0.f,0.f,0.f,0.f,0.f,0.f,0.f};
    #pragma unroll
    for (int ks = 0; ks < 8; ++ks) {
      Frag a;
      const uint4 a0 = *(const uint4*)(xrow + ks * 32);        // v=0..3
      const uint4 a1 = *(const uint4*)(xrow + ks * 32 + 16);   // v=4..7
      a.ui[0] = a0.x; a.ui[1] = a0.y; a.ui[2] = a0.z; a.ui[3] = a0.w;
      a.ui[4] = a1.x; a.ui[5] = a1.y; a.ui[6] = a1.z; a.ui[7] = a1.w;
      c = __builtin_amdgcn_wmma_f32_16x16x32_bf16(
          false, a.v, false, bfr[ks].v, (short)0, c, false, false);
    }

    // C element (vgpr r, lane) -> M = mrow + r, N = tile*16 + l15
    const int mrow = m0 + 8 * lhi;
    float* __restrict__ gp = g1   + (size_t)mrow * HIDN + tile * 16 + l15;
    float* __restrict__ ap = acc1 + (size_t)mrow * HIDN + tile * 16 + l15;
    if (m0 + 16 <= nNodes) {
      // fast path: full 16-row block, dinv rows contiguous per lane
      const float4 d0 = *(const float4*)(dinv + mrow);
      const float4 d1 = *(const float4*)(dinv + mrow + 4);
      const float dv[8] = {d0.x, d0.y, d0.z, d0.w, d1.x, d1.y, d1.z, d1.w};
      #pragma unroll
      for (int r = 0; r < 8; ++r) {
        const float val = c[r] * dv[r];
        gp[r * HIDN] = val;   // immediate-offset stores
        ap[r * HIDN] = val;   // self-loop term
      }
    } else {
      #pragma unroll
      for (int r = 0; r < 8; ++r) {
        const int m = mrow + r;
        if (m < nNodes) {
          const float val = c[r] * dinv[m];
          gp[r * HIDN] = val;
          ap[r * HIDN] = val;
        }
      }
    }
  }
}

// ---------------- edge scatter, layer 1: acc1[dst] += g1[src] ---------------
// 16 threads/edge, float4 gathers, 4 float atomics each.
__global__ void k_scatter1(const int* __restrict__ src, const int* __restrict__ dst,
                           const float* __restrict__ g1, float* __restrict__ acc1,
                           int e) {
  const long i = (long)blockIdx.x * blockDim.x + threadIdx.x;  // e*16 threads
  const long edge = i >> 4;
  const int  f4   = (int)(i & 15) * 4;
  if (edge < e) {
    const int s = src[edge], d = dst[edge];
    const float4 g = *(const float4*)(g1 + (size_t)s * HIDN + f4);
    float* a = acc1 + (size_t)d * HIDN + f4;
    atomicAdd(a + 0, g.x);
    atomicAdd(a + 1, g.y);
    atomicAdd(a + 2, g.z);
    atomicAdd(a + 3, g.w);
  }
}

// ---------------- h1 = relu(acc1 * dinv + b1), in place ---------------------
__global__ void k_relu1(float* acc1, const float* __restrict__ dinv,
                        const float* __restrict__ b1, int n) {
  const int i = blockIdx.x * blockDim.x + threadIdx.x;  // n*64 threads
  if (i < n * HIDN) {
    const int node = i >> 6, j = i & 63;
    const float v = acc1[i] * dinv[node] + b1[j];
    acc1[i] = v > 0.0f ? v : 0.0f;
  }
}

// ---------------- GEMM2 (64->2) + self loop ---------------------------------
__global__ void k_gemm2(const float* __restrict__ h1, const float* __restrict__ W2,
                        const float* __restrict__ dinv, float* __restrict__ g2,
                        float* __restrict__ acc2, int n) {
  const int node = blockIdx.x * blockDim.x + threadIdx.x;
  if (node < n) {
    float a0 = 0.0f, a1 = 0.0f;
    const float* hr = h1 + (size_t)node * HIDN;
    #pragma unroll 8
    for (int j = 0; j < HIDN; ++j) {
      const float hv = hr[j];
      a0 += hv * W2[j * EMBN + 0];
      a1 += hv * W2[j * EMBN + 1];
    }
    const float dv = dinv[node];
    a0 *= dv; a1 *= dv;
    g2[node * EMBN + 0] = a0;  acc2[node * EMBN + 0] = a0;
    g2[node * EMBN + 1] = a1;  acc2[node * EMBN + 1] = a1;
  }
}

// ---------------- edge scatter, layer 2 -------------------------------------
__global__ void k_scatter2(const int* __restrict__ src, const int* __restrict__ dst,
                           const float* __restrict__ g2, float* __restrict__ acc2,
                           int e) {
  const int i = blockIdx.x * blockDim.x + threadIdx.x;  // e*2 threads
  const int edge = i >> 1;
  const int f    = i & 1;
  if (edge < e)
    atomicAdd(&acc2[(size_t)dst[edge] * EMBN + f], g2[(size_t)src[edge] * EMBN + f]);
}

// ---------------- pooling ----------------------------------------------------
__global__ void k_zero_pool(float* sums, float* cnt, int g) {
  const int i = blockIdx.x * blockDim.x + threadIdx.x;
  if (i < g * EMBN) sums[i] = 0.0f;
  if (i < g) cnt[i] = 0.0f;
}

__global__ void k_pool(const float* __restrict__ acc2, const float* __restrict__ dinv,
                       const float* __restrict__ b2, const int* __restrict__ batch,
                       float* sums, float* cnt, int n) {
  const int node = blockIdx.x * blockDim.x + threadIdx.x;
  if (node < n) {
    const int b = batch[node];
    const float dv = dinv[node];
    atomicAdd(&sums[b * EMBN + 0], acc2[node * EMBN + 0] * dv + b2[0]);
    atomicAdd(&sums[b * EMBN + 1], acc2[node * EMBN + 1] * dv + b2[1]);
    atomicAdd(&cnt[b], 1.0f);
  }
}

__global__ void k_out(const float* __restrict__ sums, const float* __restrict__ cnt,
                      float* out, int g) {
  const int i = blockIdx.x * blockDim.x + threadIdx.x;
  if (i < g * EMBN) out[i] = sums[i] / fmaxf(cnt[i >> 1], 1.0f);
}

// ---------------------------------------------------------------------------
extern "C" void kernel_launch(void* const* d_in, const int* in_sizes, int n_in,
                              void* d_out, int out_size, void* d_ws, size_t ws_size,
                              hipStream_t stream) {
  const float* x     = (const float*)d_in[0];
  const int*   eidx  = (const int*)  d_in[1];   // [2, E] flat
  const int*   batch = (const int*)  d_in[2];
  const float* W1    = (const float*)d_in[3];
  const float* b1    = (const float*)d_in[4];
  const float* W2    = (const float*)d_in[5];
  const float* b2    = (const float*)d_in[6];

  const int N = in_sizes[2];
  const int E = in_sizes[1] / 2;
  const int G = out_size / EMBN;

  const int* src = eidx;
  const int* dst = eidx + E;

  // Workspace carving (256B aligned)
  char* ws = (char*)d_ws;
  size_t off = 0;
  auto carve = [&](size_t bytes) {
    char* p = ws + off;
    off = (off + bytes + 255) & ~(size_t)255;
    return p;
  };
  float*          dinv = (float*)carve((size_t)N * sizeof(float));      // deg -> dinv
  unsigned short* xbf  = (unsigned short*)carve((size_t)N * FIN * 2);   // bf16 x
  float*          g1   = (float*)carve((size_t)N * HIDN * sizeof(float));
  float*          acc1 = (float*)carve((size_t)N * HIDN * sizeof(float)); // -> h1
  float*          g2   = (float*)carve((size_t)N * EMBN * sizeof(float));
  float*          acc2 = (float*)carve((size_t)N * EMBN * sizeof(float));
  float*          sums = (float*)carve((size_t)G * EMBN * sizeof(float));
  float*          cnt  = (float*)carve((size_t)G * sizeof(float));
  (void)ws_size; (void)n_in;

  const int B = 256;
  auto cdiv = [](long a, long b) { return (int)((a + b - 1) / b); };

  // normalization + input conversion
  k_deg_init <<<cdiv(N, B), B, 0, stream>>>(dinv, N);
  k_deg_count<<<cdiv(E, B), B, 0, stream>>>(dst, dinv, E);
  k_dinv     <<<cdiv(N, B), B, 0, stream>>>(dinv, N);
  const long nPairs = (long)N * FIN / 2;
  k_cvt_x    <<<cdiv(nPairs, B), B, 0, stream>>>(x, (unsigned int*)xbf, nPairs);

  // layer 1: wave tasks = 4 N-tiles * ceil(N / (16*MBLK)) M-chunks, 8 waves/block
  const long tasks = 4L * cdiv(N, 16 * MBLK);
  k_gemm1    <<<cdiv(tasks, 8), B, 0, stream>>>(xbf, W1, dinv, g1, acc1, N);
  k_scatter1 <<<cdiv((long)E * 16, B), B, 0, stream>>>(src, dst, g1, acc1, E);
  k_relu1    <<<cdiv((long)N * HIDN, B), B, 0, stream>>>(acc1, dinv, b1, N);

  // layer 2
  k_gemm2    <<<cdiv(N, B), B, 0, stream>>>(acc1, W2, dinv, g2, acc2, N);
  k_scatter2 <<<cdiv((long)E * EMBN, B), B, 0, stream>>>(src, dst, g2, acc2, E);

  // pooling
  k_zero_pool<<<cdiv((long)G * EMBN, B), B, 0, stream>>>(sums, cnt, G);
  k_pool     <<<cdiv(N, B), B, 0, stream>>>(acc2, dinv, b2, batch, sums, cnt, N);
  k_out      <<<cdiv((long)G * EMBN, B), B, 0, stream>>>(sums, cnt, (float*)d_out, G);
}
